// Encoder_4097398800995
// MI455X (gfx1250) — matrix operand order repaired and checked by reading, not compiled, not run on hardware
//
#include <hip/hip_runtime.h>

// ---------------------------------------------------------------------------
// GraphSAGE encoder for MI455X (gfx1250, wave32, WMMA).
// N=100000 nodes, E=1.6M edges, D=128, L=3 layers.
//
// Memory-bound by edge aggregation (~820MB gather + ~820MB f32 atomics per
// layer, atomics L2-resident: neigh = 51MB << 192MB L2). GEMM (19.7 GFLOP)
// runs on v_wmma_f32_16x16x32_bf16 with f32 accumulation; weights pre-packed
// per call into per-lane WMMA B-fragment order (contiguous 32B/lane).
// ---------------------------------------------------------------------------

#define DD 128          // hidden dim
#define KK 256          // fused K = [h | neigh]
#define ROWS 64         // node rows per block

typedef __attribute__((ext_vector_type(16))) __bf16 v16bf;
typedef __attribute__((ext_vector_type(8)))  float  v8f;

struct __align__(16) U4 { unsigned int x, y, z, w; };
struct __align__(32) U4x2 { U4 lo, hi; };
struct __align__(8)  US4 { unsigned short a, b, c, d; };

__device__ __forceinline__ unsigned short f32_to_bf16(float f) {
    unsigned int u = __float_as_uint(f);
    u += 0x7FFFu + ((u >> 16) & 1u);     // round-to-nearest-even
    return (unsigned short)(u >> 16);
}

// --- deg[dst] += 1 ----------------------------------------------------------
__global__ void deg_kernel(const int* __restrict__ dst, float* __restrict__ deg, int E) {
    int i = blockIdx.x * blockDim.x + threadIdx.x;
    if (i < E) atomicAdd(&deg[dst[i]], 1.0f);
}

// --- deg -> 1/max(deg,1) in place ------------------------------------------
__global__ void invdeg_kernel(float* __restrict__ deg, int N) {
    int i = blockIdx.x * blockDim.x + threadIdx.x;
    if (i < N) deg[i] = 1.0f / fmaxf(deg[i], 1.0f);
}

// --- pack [Ws;Wn] (f32, KxN row-major) into WMMA bf16 B-fragment order ------
// packed[((ct*8+ks)*32 + lane)*16 + i] = bf16( Wcat[32*ks + 16*(lane>>4) + i][16*ct + (lane&15)] )
__global__ void pack_kernel(const float* __restrict__ Ws, const float* __restrict__ Wn,
                            unsigned short* __restrict__ packW, int L) {
    int p = blockIdx.x * blockDim.x + threadIdx.x;
    const int perLayer = 8 * 8 * 32 * 16;          // 32768
    if (p >= L * perLayer) return;
    int layer = p / perLayer;
    int rem   = p - layer * perLayer;
    int i    = rem & 15;
    int lane = (rem >> 4) & 31;
    int ks   = (rem >> 9) & 7;
    int ct   = rem >> 12;
    int n = (lane & 15) + 16 * ct;
    int k = 32 * ks + 16 * (lane >> 4) + i;
    float v = (k < DD) ? Ws[(size_t)layer * DD * DD + (size_t)k * DD + n]
                       : Wn[(size_t)layer * DD * DD + (size_t)(k - DD) * DD + n];
    packW[p] = f32_to_bf16(v);
}

// --- h = emb[x]  (one thread per (node, 4 cols)) ----------------------------
__global__ void embed_kernel(const float* __restrict__ emb, const int* __restrict__ x,
                             float* __restrict__ h, int N) {
    int t = blockIdx.x * blockDim.x + threadIdx.x;
    if (t >= N * 32) return;
    int node = t >> 5;
    int c = (t & 31) * 4;
    const float4 v = *(const float4*)(emb + (size_t)x[node] * DD + c);
    *(float4*)(h + (size_t)node * DD + c) = v;
}

// --- neigh[dst] += h[src]  (one thread per (edge, 4 cols), f32 atomics) -----
__global__ void scatter_kernel(const int* __restrict__ src, const int* __restrict__ dst,
                               const float* __restrict__ h, float* __restrict__ neigh, int E) {
    int t = blockIdx.x * blockDim.x + threadIdx.x;
    if (t >= E * 32) return;
    int e = t >> 5;
    int c = (t & 31) * 4;
    int s = src[e], d = dst[e];
    const float4 v = *(const float4*)(h + (size_t)s * DD + c);
    float* o = neigh + (size_t)d * DD + c;
    atomicAdd(o + 0, v.x);
    atomicAdd(o + 1, v.y);
    atomicAdd(o + 2, v.z);
    atomicAdd(o + 3, v.w);
}

// --- fused: h = l2norm(relu([h|neigh*invdeg] @ [Ws;Wn] + b)) + emb[x] -------
// Block: 256 threads (8 waves), 64 node rows x 128 cols. In-place on h
// (each block only touches its own 64 rows).
__global__ void __launch_bounds__(256)
sage_gemm_kernel(float* __restrict__ h, const float* __restrict__ neigh,
                 const float* __restrict__ invdeg,
                 const unsigned short* __restrict__ packW,  // this layer's 32768 bf16
                 const float* __restrict__ bias,            // this layer's 128 floats
                 const float* __restrict__ emb, const int* __restrict__ x, int N) {
    __shared__ __align__(16) unsigned char smem[ROWS * KK * 2 + ROWS * DD * 4]; // 32KB + 32KB
    unsigned short* Alds = (unsigned short*)smem;                 // [64][256] bf16
    float*          Hlds = (float*)(smem + ROWS * KK * 2);        // [64][128] f32
    float*          Slds = (float*)smem;                          // aliases Alds (post-K reuse)

    const int tid  = threadIdx.x;
    const int lane = tid & 31;
    const int w    = tid >> 5;
    const int blockRow = blockIdx.x * ROWS;

    // ---- stage A = [h | neigh*invdeg] as bf16 into LDS ----
    for (int idx = tid; idx < ROWS * (KK / 4); idx += 256) {
        int row = idx >> 6;          // 0..63
        int k   = (idx & 63) * 4;    // 0..252
        int g   = blockRow + row;
        float4 v = make_float4(0.f, 0.f, 0.f, 0.f);
        if (g < N) {
            if (k < DD) {
                v = *(const float4*)(h + (size_t)g * DD + k);
            } else {
                v = *(const float4*)(neigh + (size_t)g * DD + (k - DD));
                float s = invdeg[g];
                v.x *= s; v.y *= s; v.z *= s; v.w *= s;
            }
        }
        US4 u; u.a = f32_to_bf16(v.x); u.b = f32_to_bf16(v.y);
        u.c = f32_to_bf16(v.z); u.d = f32_to_bf16(v.w);
        *(US4*)(Alds + (size_t)row * KK + k) = u;
    }
    __syncthreads();

    // ---- WMMA main loop: wave w -> row-tile rt = w>>1, col-half ch = w&1 ----
    const int rt   = w >> 1;         // 0..3 (16 rows each)
    const int ch   = w & 1;          // 0..1 (4 col-tiles each)
    const int half = lane >> 4;
    const int m    = lane & 15;
    const unsigned short* Abase = Alds + (size_t)(rt * 16 + m) * KK;

    v8f acc[4];
    #pragma unroll
    for (int ct = 0; ct < 4; ++ct)
        #pragma unroll
        for (int e = 0; e < 8; ++e) acc[ct][e] = 0.0f;

    #pragma unroll
    for (int ks = 0; ks < 8; ++ks) {
        const int kb = ks * 32;
        // A fragment: elems 0..15 = K kb+16*half+i
        U4x2 aq;
        aq.lo = *(const U4*)(Abase + kb + 16 * half);
        aq.hi = *(const U4*)(Abase + kb + 16 * half + 8);
        const v16bf av = __builtin_bit_cast(v16bf, aq);
        #pragma unroll
        for (int ct = 0; ct < 4; ++ct) {
            const int ctAbs = ch * 4 + ct;
            const unsigned short* bp = packW + ((size_t)(ctAbs * 8 + ks) * 32 + lane) * 16;
            U4x2 bq;
            bq.lo = *(const U4*)(bp);
            bq.hi = *(const U4*)(bp + 8);
            const v16bf bv = __builtin_bit_cast(v16bf, bq);
            acc[ct] = __builtin_amdgcn_wmma_f32_16x16x32_bf16(
                false, av, false, bv, (short)0, acc[ct], false, false);
        }
    }

    // ---- epilogue: bias + relu -> Hlds ----
    #pragma unroll
    for (int ct = 0; ct < 4; ++ct) {
        const int ctAbs = ch * 4 + ct;
        const int col = ctAbs * 16 + (lane & 15);
        const float bcol = bias[col];
        #pragma unroll
        for (int r = 0; r < 8; ++r) {
            int row = rt * 16 + (lane >> 4) * 8 + r;
            float v = acc[ct][r] + bcol;
            Hlds[(size_t)row * DD + col] = fmaxf(v, 0.0f);
        }
    }
    __syncthreads();   // all Alds reads done; Hlds complete; Slds reuse safe

    // ---- l2 normalize + residual, write h in place ----
    {
        const int row = tid >> 2;       // 0..63
        const int q   = tid & 3;        // 32-col quarter
        const float* hr = Hlds + (size_t)row * DD + q * 32;
        float ss = 0.0f;
        #pragma unroll
        for (int j = 0; j < 8; ++j) {
            float4 v = *(const float4*)(hr + j * 4);
            ss += v.x * v.x + v.y * v.y + v.z * v.z + v.w * v.w;
        }
        Slds[row * 4 + q] = ss;
    }
    __syncthreads();
    {
        const int row = tid >> 2;
        const int q   = tid & 3;
        const int g   = blockRow + row;
        float tot = Slds[row * 4 + 0] + Slds[row * 4 + 1]
                  + Slds[row * 4 + 2] + Slds[row * 4 + 3];
        float scale = 1.0f / fmaxf(__builtin_sqrtf(tot), 1e-12f);
        if (g < N) {
            const int xv = x[g];
            const float* hr = Hlds + (size_t)row * DD + q * 32;
            const float* er = emb + (size_t)xv * DD + q * 32;
            float* out = h + (size_t)g * DD + q * 32;
            #pragma unroll
            for (int j = 0; j < 8; ++j) {
                float4 v = *(const float4*)(hr + j * 4);
                float4 e = *(const float4*)(er + j * 4);
                v.x = v.x * scale + e.x;
                v.y = v.y * scale + e.y;
                v.z = v.z * scale + e.z;
                v.w = v.w * scale + e.w;
                *(float4*)(out + j * 4) = v;
            }
        }
    }
}

// ---------------------------------------------------------------------------
extern "C" void kernel_launch(void* const* d_in, const int* in_sizes, int n_in,
                              void* d_out, int out_size, void* d_ws, size_t ws_size,
                              hipStream_t stream) {
    const int*   x   = (const int*)d_in[0];
    const int*   src = (const int*)d_in[1];
    const int*   dst = (const int*)d_in[2];
    const float* emb = (const float*)d_in[3];
    const float* Ws  = (const float*)d_in[4];
    const float* Wn  = (const float*)d_in[5];
    const float* b   = (const float*)d_in[6];

    const int N = in_sizes[0];
    const int E = in_sizes[1];
    const int L = in_sizes[6] / DD;

    float* h = (float*)d_out;                 // h lives in d_out, updated in place

    char* ws = (char*)d_ws;
    float* neigh = (float*)ws;                                    // N*D f32
    float* deg   = (float*)(ws + (size_t)N * DD * 4);             // N f32 (becomes invdeg)
    unsigned short* packW =
        (unsigned short*)(ws + (size_t)N * DD * 4 + (size_t)N * 4);  // L*32768 bf16

    // degree + inverse degree
    hipMemsetAsync(deg, 0, (size_t)N * 4, stream);
    deg_kernel<<<(E + 255) / 256, 256, 0, stream>>>(dst, deg, E);
    invdeg_kernel<<<(N + 255) / 256, 256, 0, stream>>>(deg, N);

    // pack weights to bf16 WMMA B-fragment layout (once per call, all layers)
    pack_kernel<<<(L * 32768 + 255) / 256, 256, 0, stream>>>(Ws, Wn, packW, L);

    // h = emb[x]
    embed_kernel<<<(N * 32 + 255) / 256, 256, 0, stream>>>(emb, x, h, N);

    for (int l = 0; l < L; ++l) {
        hipMemsetAsync(neigh, 0, (size_t)N * DD * 4, stream);
        scatter_kernel<<<(E * 32 + 255) / 256, 256, 0, stream>>>(src, dst, h, neigh, E);
        sage_gemm_kernel<<<(N + ROWS - 1) / ROWS, 256, 0, stream>>>(
            h, neigh, deg, packW + (size_t)l * 32768, b + (size_t)l * DD, emb, x, N);
    }
}
